// BahdanauAttention_25220047962265
// MI455X (gfx1250) — matrix operand order repaired and checked by reading, not compiled, not run on hardware
//
#include <hip/hip_runtime.h>
#include <hip/hip_bf16.h>

// ---------------------------------------------------------------------------
// Bahdanau attention, fused for MI455X (gfx1250, wave32, WMMA bf16)
// S=2048 B=32 QD=VD=E=1024
// ---------------------------------------------------------------------------

#define S_LEN 2048
#define B_SZ  32
#define E_SZ  1024
#define VD_SZ 1024
#define NROWS (S_LEN * B_SZ)          // 65536 rows of (s,b)

typedef __attribute__((ext_vector_type(16))) __bf16        v16bf;
typedef __attribute__((ext_vector_type(8)))  float         v8f;
typedef __attribute__((ext_vector_type(8)))  unsigned int  v8u;

union ABits { v8u u; v16bf bf; };

// pack two f32 into one dword of bf16 (truncation) with a single v_perm_b32
__device__ __forceinline__ unsigned int pack_bf16(float hi, float lo) {
    return __builtin_amdgcn_perm(__float_as_uint(hi), __float_as_uint(lo), 0x07060302u);
}

// ---------------------------------------------------------------------------
// 1) Wv (f32 [E,VD] row-major) -> bf16, pre-swizzled into WMMA B-matrix order:
//    dest index = (((ntile*32 + kstep)*32 + lane)*16 + elem) bf16 elements
//    lanes 0-15: N=lane,    K=kstep*32 + elem
//    lanes16-31: N=lane-16, K=kstep*32 + 16 + elem
// ---------------------------------------------------------------------------
__global__ void k_cvt_wv(const float* __restrict__ wv, unsigned short* __restrict__ wvb) {
    int idx = blockIdx.x * 256 + threadIdx.x;     // 0 .. 1048575
    int n = idx >> 10, k = idx & 1023;
    unsigned int bits = __float_as_uint(wv[idx]);
    bits += 0x7FFFu + ((bits >> 16) & 1u);        // round-to-nearest-even
    int nt = n >> 4, nl = n & 15, ks = k >> 5, ki = k & 31;
    int lane = nl + ((ki >= 16) ? 16 : 0);
    int elem = ki & 15;
    wvb[(((nt * 32 + ks) * 32 + lane) << 4) + elem] = (unsigned short)(bits >> 16);
}

// ---------------------------------------------------------------------------
// 2a) pq[b][e] = query[b,:] . Wq[e,:]   (32x1024 outputs, tiny)
// ---------------------------------------------------------------------------
__global__ void k_pq(const float* __restrict__ query, const float* __restrict__ wq,
                     float* __restrict__ pq) {
    int idx = blockIdx.x * 256 + threadIdx.x;     // 0 .. 32767
    int bq = idx >> 10, e = idx & 1023;
    const float4* qp = (const float4*)(query + (size_t)bq * 1024);
    const float4* wp = (const float4*)(wq + (size_t)e * 1024);
    float acc = 0.f;
    for (int i = 0; i < 256; ++i) {
        float4 a = qp[i], w = wp[i];
        acc += a.x * w.x + a.y * w.y + a.z * w.z + a.w * w.w;
    }
    pq[idx] = acc;
}

// ---------------------------------------------------------------------------
// 2b) nv[e] = g * v[e] / ||v||
// ---------------------------------------------------------------------------
__global__ void k_nv(const float* __restrict__ v, const float* __restrict__ g,
                     float* __restrict__ nv) {
    __shared__ float red[256];
    int tid = threadIdx.x;
    float s = 0.f;
    for (int i = tid; i < E_SZ; i += 256) { float x = v[i]; s += x * x; }
    red[tid] = s; __syncthreads();
    for (int o = 128; o > 0; o >>= 1) {
        if (tid < o) red[tid] += red[tid + o];
        __syncthreads();
    }
    float scale = g[0] * rsqrtf(red[0]);
    for (int i = tid; i < E_SZ; i += 256) nv[i] = v[i] * scale;
}

// ---------------------------------------------------------------------------
// 3) Main fused kernel: scores[r] = sum_e tanh(value[r,:]·Wv[e,:] + pq[b(r),e]
//                                            + bias[e]) * nv[e]
//    One wave per 16-row M-tile, bf16 WMMA 16x16x32, f32 accumulate.
//    N (=E) swept in 8 groups of 8 tiles; A converted f32->bf16 in-register.
// ---------------------------------------------------------------------------
__global__ void __launch_bounds__(256)
k_scores(const float* __restrict__ value, const unsigned short* __restrict__ wvb,
         const float* __restrict__ pq, const float* __restrict__ nv,
         const float* __restrict__ bias, float* __restrict__ scores) {
    const int lane  = threadIdx.x & 31;
    const int wave  = threadIdx.x >> 5;
    const int mtile = blockIdx.x * 8 + wave;          // 4096 M-tiles
    const int rbase = mtile * 16;                     // first flat row (s*B+b)
    const int nlane = lane & 15;
    const int arow  = rbase + nlane;                  // value row this lane feeds A with
    const int koff  = (lane >= 16) ? 16 : 0;          // A layout half-wave K offset
    const float* aptr = value + (size_t)arow * VD_SZ + koff;
    const v8u* bbase = (const v8u*)wvb;
    const int rk = (lane >> 4) * 8;                   // C rows: 0..7 (lo half) / 8..15 (hi half)

    float sacc[8];
#pragma unroll
    for (int i = 0; i < 8; ++i) sacc[i] = 0.f;

    for (int ng = 0; ng < 8; ++ng) {                  // 8 N-groups x 8 N-tiles = 64
        v8f c[8];
#pragma unroll
        for (int t = 0; t < 8; ++t) c[t] = (v8f){0.f,0.f,0.f,0.f,0.f,0.f,0.f,0.f};

        for (int ks = 0; ks < 32; ++ks) {             // K = 1024 in steps of 32
            // --- load 16 f32 of this lane's A fragment, pack to bf16 ---
            const float* p = aptr + ks * 32;
            float4 f0 = *(const float4*)(p);
            float4 f1 = *(const float4*)(p + 4);
            float4 f2 = *(const float4*)(p + 8);
            float4 f3 = *(const float4*)(p + 12);
            ABits a;
            a.u[0] = pack_bf16(f0.y, f0.x);
            a.u[1] = pack_bf16(f0.w, f0.z);
            a.u[2] = pack_bf16(f1.y, f1.x);
            a.u[3] = pack_bf16(f1.w, f1.z);
            a.u[4] = pack_bf16(f2.y, f2.x);
            a.u[5] = pack_bf16(f2.w, f2.z);
            a.u[6] = pack_bf16(f3.y, f3.x);
            a.u[7] = pack_bf16(f3.w, f3.z);

#pragma unroll
            for (int t = 0; t < 8; ++t) {             // reuse A across 8 N-tiles
                int nt = ng * 8 + t;
                ABits bb;
                bb.u = bbase[(nt * 32 + ks) * 32 + lane];
                c[t] = __builtin_amdgcn_wmma_f32_16x16x32_bf16(
                    /*neg_a=*/false, a.bf, /*neg_b=*/false, bb.bf,
                    /*c_mod=*/(short)0, c[t], /*reuse_a=*/false, /*reuse_b=*/false);
            }
        }

        // --- fused epilogue: + pq + bias, tanh, dot with nv ---
#pragma unroll
        for (int t = 0; t < 8; ++t) {
            int n = (ng * 8 + t) * 16 + nlane;
            float nvn = nv[n];
            float bn  = bias[n];
#pragma unroll
            for (int k = 0; k < 8; ++k) {
                int row = rbase + rk + k;             // row = s*32 + b -> b = row & 31
                float x = c[t][k] + pq[(row & 31) * E_SZ + n] + bn;
                sacc[k] += tanhf(x) * nvn;
            }
        }
    }

    // cross-lane reduce over the 16 N-lanes of each half-wave
#pragma unroll
    for (int k = 0; k < 8; ++k) {
        float s = sacc[k];
        s += __shfl_xor(s, 1);
        s += __shfl_xor(s, 2);
        s += __shfl_xor(s, 4);
        s += __shfl_xor(s, 8);
        if (nlane == 0) scores[rbase + rk + k] = s;
    }
}

// ---------------------------------------------------------------------------
// 4) Softmax over S per batch column b; writes attn to ws and both d_out copies
// ---------------------------------------------------------------------------
__global__ void k_softmax(const float* __restrict__ scores,
                          const unsigned char* __restrict__ mask,
                          float* __restrict__ attn, float* __restrict__ out) {
    const int b = blockIdx.x, tid = threadIdx.x;
    __shared__ float red[256];

    float m = -3.0e38f;
    for (int s = tid; s < S_LEN; s += 256) {
        int i = s * B_SZ + b;
        float x = mask[i] ? -3.0e38f : scores[i];
        m = fmaxf(m, x);
    }
    red[tid] = m; __syncthreads();
    for (int o = 128; o > 0; o >>= 1) {
        if (tid < o) red[tid] = fmaxf(red[tid], red[tid + o]);
        __syncthreads();
    }
    m = red[0]; __syncthreads();

    float sum = 0.f;
    for (int s = tid; s < S_LEN; s += 256) {
        int i = s * B_SZ + b;
        if (!mask[i]) sum += __expf(scores[i] - m);
    }
    red[tid] = sum; __syncthreads();
    for (int o = 128; o > 0; o >>= 1) {
        if (tid < o) red[tid] += red[tid + o];
        __syncthreads();
    }
    float inv = 1.0f / red[0];

    float* o0 = out + B_SZ * VD_SZ;                   // attn copy 1
    float* o1 = o0 + S_LEN * B_SZ;                    // attn copy 2
    for (int s = tid; s < S_LEN; s += 256) {
        int i = s * B_SZ + b;
        float a = mask[i] ? 0.f : __expf(scores[i] - m) * inv;
        attn[i] = a; o0[i] = a; o1[i] = a;
    }
}

// ---------------------------------------------------------------------------
// 5) context[b][d] = sum_s attn[s][b] * value[s][b][d]  (split-S, no atomics)
// ---------------------------------------------------------------------------
__global__ void k_ctx_partial(const float* __restrict__ value,
                              const float* __restrict__ attn,
                              float* __restrict__ partial) {
    const int b = blockIdx.x, sc = blockIdx.y, tid = threadIdx.x;
    const int d0 = tid * 4;
    float4 acc = {0.f, 0.f, 0.f, 0.f};
    for (int s = sc * 128; s < sc * 128 + 128; ++s) {
        int r = s * B_SZ + b;
        float a = attn[r];
        float4 vv = *(const float4*)(value + ((size_t)r << 10) + d0);
        acc.x += a * vv.x; acc.y += a * vv.y; acc.z += a * vv.z; acc.w += a * vv.w;
    }
    *(float4*)(partial + ((size_t)(b * 16 + sc) << 10) + d0) = acc;
}

__global__ void k_ctx_reduce(const float* __restrict__ partial, float* __restrict__ out) {
    int idx = blockIdx.x * 256 + threadIdx.x;         // 0 .. 32767
    int b = idx >> 10, d = idx & 1023;
    float s = 0.f;
#pragma unroll
    for (int sc = 0; sc < 16; ++sc) s += partial[((b * 16 + sc) << 10) + d];
    out[idx] = s;
}

// ---------------------------------------------------------------------------
extern "C" void kernel_launch(void* const* d_in, const int* in_sizes, int n_in,
                              void* d_out, int out_size, void* d_ws, size_t ws_size,
                              hipStream_t stream) {
    const float*         query = (const float*)d_in[0];
    const float*         value = (const float*)d_in[1];
    const unsigned char* mask  = (const unsigned char*)d_in[2];
    const float*         Wq    = (const float*)d_in[3];
    const float*         Wv    = (const float*)d_in[4];
    const float*         v     = (const float*)d_in[5];
    const float*         bias  = (const float*)d_in[6];
    const float*         g     = (const float*)d_in[7];
    float* out = (float*)d_out;
    char*  ws  = (char*)d_ws;

    // scratch layout (all 4KB-aligned offsets), total ~4.7 MB
    unsigned short* wvb     = (unsigned short*)(ws);                              // 2 MB bf16 swizzled Wv
    float*          pq      = (float*)(ws + (2u << 20));                          // 128 KB
    float*          nv      = (float*)(ws + (2u << 20) + (128u << 10));           // 4 KB
    float*          scores  = (float*)(ws + (2u << 20) + (132u << 10));           // 256 KB
    float*          attn    = (float*)(ws + (2u << 20) + (388u << 10));           // 256 KB
    float*          partial = (float*)(ws + (2u << 20) + (644u << 10));           // 2 MB

    k_cvt_wv   <<<4096, 256, 0, stream>>>(Wv, wvb);
    k_pq       <<<128,  256, 0, stream>>>(query, Wq, pq);
    k_nv       <<<1,    256, 0, stream>>>(v, g, nv);
    k_scores   <<<512,  256, 0, stream>>>(value, wvb, pq, nv, bias, scores);
    k_softmax  <<<32,   256, 0, stream>>>(scores, mask, attn, out);
    k_ctx_partial<<<dim3(32, 16), 256, 0, stream>>>(value, attn, partial);
    k_ctx_reduce <<<128, 256, 0, stream>>>(partial, out);
}
